// AUGRU_Cell_29214367547613
// MI455X (gfx1250) — matrix operand order repaired and checked by reading, not compiled, not run on hardware
//
#include <hip/hip_runtime.h>
#include <hip/hip_bf16.h>

#define D 256

typedef __attribute__((ext_vector_type(16))) __bf16 v16bf;
typedef __attribute__((ext_vector_type(8)))  __bf16 v8bf;
typedef __attribute__((ext_vector_type(4)))  __bf16 v4bf;
typedef __attribute__((ext_vector_type(8)))  float  v8f;
typedef __attribute__((ext_vector_type(4)))  float  v4f;

union ABfrag {
    v16bf v;
    v8bf  h[2];
    v4f   f4[2];
};

static __device__ inline v8f wmma_bf16(v16bf A, v16bf Bm, v8f C) {
    // D = A(16x32 bf16) * B(32x16 bf16) + C(16x16 f32)
    return __builtin_amdgcn_wmma_f32_16x16x32_bf16(
        /*neg_a=*/false, A, /*neg_b=*/false, Bm,
        /*c_mod=*/(short)0, C, /*reuse_a=*/false, /*reuse_b=*/false);
}

// ---------------------------------------------------------------------------
// Pre-pass: convert 6 fp32 weight matrices [256x256] (row-major, K x N) into
// bf16, transposed to N-major / K-contiguous so each WMMA B-fragment is a
// pair of 16-byte reads per lane.  Output layout: WT[m][n][k], m in 0..5.
// ---------------------------------------------------------------------------
__global__ __launch_bounds__(256) void augru_pack_weights(
    const float* __restrict__ Wu, const float* __restrict__ Uu,
    const float* __restrict__ Wr, const float* __restrict__ Ur,
    const float* __restrict__ Wh, const float* __restrict__ Uh,
    __bf16* __restrict__ WT)
{
    const float* Ws[6] = {Wu, Uu, Wr, Ur, Wh, Uh};
    int idx = blockIdx.x * 256 + threadIdx.x;      // 6 * 65536 total
    int m = idx >> 16;
    int e = idx & 65535;
    int n = e >> 8;
    int k = e & 255;
    WT[idx] = (__bf16)Ws[m][k * D + n];
}

// ---------------------------------------------------------------------------
// Fused AUGRU cell.
//   Workgroup: 256 threads = 8 wave32, tile = 32 rows x 256 cols.
//   Wave w owns N columns [w*32, w*32+32) and BOTH 16-row M-subtiles, so each
//   B fragment (weight tile) is reused by two WMMAs -> halves L2 weight reads.
//   Per wave: 2 mt x 2 nt x 4 accumulator types = 16 x v8f accumulators.
// ---------------------------------------------------------------------------
__global__ __launch_bounds__(256, 2) void augru_cell_kernel(
    const float* __restrict__ x,  const float* __restrict__ h1,
    const float* __restrict__ att,
    const __bf16* __restrict__ WT,           // 6 packed bf16 matrices
    const float* __restrict__ bu, const float* __restrict__ br,
    const float* __restrict__ bh,
    float* __restrict__ out)
{
    __shared__ __bf16 xs[32][272];           // 544 B stride (16B aligned)
    __shared__ __bf16 hs[32][272];
    __shared__ float  hf[32][260];           // fp32 h1 for epilogue (1040 B stride)

    const int tid  = threadIdx.x;
    const int lane = tid & 31;
    const int wid  = tid >> 5;
    const int half = lane >> 4;              // 0 / 1
    const int l16  = lane & 15;
    const int rowBase = blockIdx.x * 32;

    // ---- Stage x / h tiles (32x256 f32 -> bf16 + f32 LDS), streaming loads ----
    for (int it = 0; it < 8; ++it) {
        int f4 = it * 256 + tid;             // 0..2047 float4 slots
        int r  = f4 >> 6;                    // 64 float4 per row
        int c  = (f4 & 63) << 2;
        v4f vx = __builtin_nontemporal_load((const v4f*)(x  + (size_t)(rowBase + r) * D + c));
        v4f vh = __builtin_nontemporal_load((const v4f*)(h1 + (size_t)(rowBase + r) * D + c));
        v4bf bx, bhh;
        for (int j = 0; j < 4; ++j) { bx[j] = (__bf16)vx[j]; bhh[j] = (__bf16)vh[j]; }
        *(v4bf*)&xs[r][c] = bx;
        *(v4bf*)&hs[r][c] = bhh;
        *(v4f*)&hf[r][c]  = vh;
    }
    __syncthreads();

    const int nw = wid * 32;                 // wave's N base (2 tiles of 16)

    const __bf16* Wt[6];
    for (int m = 0; m < 6; ++m) Wt[m] = WT + (size_t)m * (D * D);

    // Warm L2 path for this wave's weight columns (global_prefetch_b8)
    for (int m = 0; m < 6; ++m)
        __builtin_prefetch(Wt[m] + (size_t)(nw + l16) * D, 0, 1);

    v8f acc_u[2][2], acc_r[2][2], acc_x[2][2], acc_h[2][2];
    for (int t = 0; t < 2; ++t)
        for (int mt = 0; mt < 2; ++mt) {
            acc_u[t][mt] = {}; acc_r[t][mt] = {};
            acc_x[t][mt] = {}; acc_h[t][mt] = {};
        }

    // ---- K loop: 8 blocks of K=32 ----
    for (int kb = 0; kb < 8; ++kb) {
        // A fragments (16-bit A 16x32 layout): two aligned 16B LDS chunks each,
        // for both 16-row M-subtiles.
        const int ka = kb * 32 + half * 8;
        ABfrag ax0, ax1, ah0, ah1;
        ax0.h[0] = *(const v8bf*)&xs[l16][ka];
        ax0.h[1] = *(const v8bf*)&xs[l16][ka + 16];
        ah0.h[0] = *(const v8bf*)&hs[l16][ka];
        ah0.h[1] = *(const v8bf*)&hs[l16][ka + 16];
        ax1.h[0] = *(const v8bf*)&xs[16 + l16][ka];
        ax1.h[1] = *(const v8bf*)&xs[16 + l16][ka + 16];
        ah1.h[0] = *(const v8bf*)&hs[16 + l16][ka];
        ah1.h[1] = *(const v8bf*)&hs[16 + l16][ka + 16];

        for (int t = 0; t < 2; ++t) {
            // B fragment: lane holds column (nw+t*16+l16), K = kb*32+half*8+e (e<8)
            // and kb*32+16+half*8+(e-8) (e>=8) in the transposed bf16 weights -> 2 x b128 loads each.
            const size_t boff = (size_t)(nw + t * 16 + l16) * D + kb * 32 + half * 8;
            ABfrag b0, b1, b2, b3, b4, b5;
#define LDB(bb, m) { bb.f4[0] = *(const v4f*)(Wt[m] + boff); \
                     bb.f4[1] = *(const v4f*)(Wt[m] + boff + 16); }
            // Load all six weight fragments up front (distinct registers) so
            // WMMAs retire against a draining loadcnt instead of waiting to 0.
            LDB(b0, 0); LDB(b2, 2); LDB(b4, 4);   // x-side weights Wu, Wr, Wh
            LDB(b1, 1); LDB(b3, 3); LDB(b5, 5);   // h-side weights Uu, Ur, Uh
#undef LDB
            // Each B fragment feeds both M-subtiles; accumulator chains are
            // interleaved (distance >= 4) to cover XDL accumulation latency.
            acc_u[t][0] = wmma_bf16(ax0.v, b0.v, acc_u[t][0]);
            acc_u[t][1] = wmma_bf16(ax1.v, b0.v, acc_u[t][1]);
            acc_r[t][0] = wmma_bf16(ax0.v, b2.v, acc_r[t][0]);
            acc_r[t][1] = wmma_bf16(ax1.v, b2.v, acc_r[t][1]);
            acc_x[t][0] = wmma_bf16(ax0.v, b4.v, acc_x[t][0]);
            acc_x[t][1] = wmma_bf16(ax1.v, b4.v, acc_x[t][1]);
            acc_u[t][0] = wmma_bf16(ah0.v, b1.v, acc_u[t][0]);
            acc_u[t][1] = wmma_bf16(ah1.v, b1.v, acc_u[t][1]);
            acc_r[t][0] = wmma_bf16(ah0.v, b3.v, acc_r[t][0]);
            acc_r[t][1] = wmma_bf16(ah1.v, b3.v, acc_r[t][1]);
            acc_h[t][0] = wmma_bf16(ah0.v, b5.v, acc_h[t][0]);
            acc_h[t][1] = wmma_bf16(ah1.v, b5.v, acc_h[t][1]);
        }
    }

    // ---- Epilogue: gates + attention blend (C/D layout: M=v+8*half, N=l16) ----
    for (int mt = 0; mt < 2; ++mt) {
        const int rowT = rowBase + mt * 16 + half * 8;
        const int rloc = mt * 16 + half * 8;
        float av[8];
        for (int v = 0; v < 8; ++v) av[v] = att[rowT + v];

        for (int t = 0; t < 2; ++t) {
            const int col = nw + t * 16 + l16;
            const float bub = bu[col], brb = br[col], bhb = bh[col];
            for (int v = 0; v < 8; ++v) {
                const int row = rowT + v;
                const float u  = 1.0f / (1.0f + __expf(-(acc_u[t][mt][v] + bub)));
                const float r  = 1.0f / (1.0f + __expf(-(acc_r[t][mt][v] + brb)));
                const float hh = tanhf(acc_x[t][mt][v] + r * acc_h[t][mt][v] + bhb);
                const float uh = av[v] * u;
                const float hp = hf[rloc + v][col];          // fp32 h1 from LDS
                __builtin_nontemporal_store((1.0f - uh) * hp + uh * hh,
                                            out + (size_t)row * D + col);
            }
        }
    }
}

extern "C" void kernel_launch(void* const* d_in, const int* in_sizes, int n_in,
                              void* d_out, int out_size, void* d_ws, size_t ws_size,
                              hipStream_t stream) {
    const float* x  = (const float*)d_in[0];
    const float* h1 = (const float*)d_in[1];
    const float* a  = (const float*)d_in[2];
    const float* Wu = (const float*)d_in[3];
    const float* Uu = (const float*)d_in[4];
    const float* bu = (const float*)d_in[5];
    const float* Wr = (const float*)d_in[6];
    const float* Ur = (const float*)d_in[7];
    const float* br = (const float*)d_in[8];
    const float* Wh = (const float*)d_in[9];
    const float* Uh = (const float*)d_in[10];
    const float* bh = (const float*)d_in[11];

    __bf16* WT = (__bf16*)d_ws;              // 6 * 256*256 bf16 = 768 KB

    augru_pack_weights<<<(6 * D * D) / 256, 256, 0, stream>>>(Wu, Uu, Wr, Ur, Wh, Uh, WT);

    const int rows   = in_sizes[0] / D;      // 65536
    const int blocks = rows / 32;            // 2048 workgroups
    augru_cell_kernel<<<blocks, 256, 0, stream>>>(x, h1, a, WT, bu, br, bh, (float*)d_out);
}
